// V2V_AttFusion_5016521802206
// MI455X (gfx1250) — compile-verified, matrix operand order checked by reading, not run on hardware
//
#include <hip/hip_runtime.h>

#define C_  64
#define H_  128
#define W_  128
#define HW_ (H_*W_)
#define L_  5
#define B_  2
#define N_  10
#define NEG_ (-1e30f)

typedef __attribute__((ext_vector_type(16))) __bf16 v16bf;
typedef __attribute__((ext_vector_type(8)))  float  v8f;

union ABf {
    v16bf v;
    unsigned short u[16];
    uint4 q[2];
};

__device__ __forceinline__ unsigned short f2bf(float f) {
    unsigned int x = __float_as_uint(f);
    unsigned int r = x + 0x7FFFu + ((x >> 16) & 1u);
    return (unsigned short)(r >> 16);
}
__device__ __forceinline__ float bf2f(unsigned short h) {
    return __uint_as_float(((unsigned int)h) << 16);
}
__device__ __forceinline__ v8f vzero() {
    v8f z = {0.f,0.f,0.f,0.f,0.f,0.f,0.f,0.f};
    return z;
}
// Load A/B fragment rows that are contiguous in K: elements [k0..k0+7] and [k0+16..k0+23]
__device__ __forceinline__ void load_a16(ABf& f, const unsigned short* row, int k0) {
    f.q[0] = *(const uint4*)(row + k0);
    f.q[1] = *(const uint4*)(row + k0 + 16);
}
__device__ __forceinline__ v8f wmma_bf16(const ABf& a, const ABf& b, v8f c) {
    return __builtin_amdgcn_wmma_f32_16x16x32_bf16(false, a.v, false, b.v, (short)0, c, false, false);
}

// ---------------------------------------------------------------------------
// Kernel A: 1x1 conv + bias + BN + ReLU  ->  bf16 NCHW output
// grid = (HW/128, N), block = 256 (8 waves). Each wave: 16 positions, all 64 outs.
// ---------------------------------------------------------------------------
__global__ __launch_bounds__(256) void cbr_kernel(const float* __restrict__ x,
                                                  const int*   __restrict__ rl,
                                                  const float* __restrict__ w,
                                                  const float* __restrict__ bias,
                                                  const float* __restrict__ bn,
                                                  unsigned short* __restrict__ out,
                                                  int useEgo) {
    __shared__ __align__(16) unsigned short Wl[C_*C_];   // [o][c], BN-scaled
    __shared__ __align__(16) unsigned short Xl[128*C_];  // [p][c]
    int n  = blockIdx.y;
    int p0 = blockIdx.x * 128;
    int t  = threadIdx.x;

    int src = n;
    if (useEgo) {
        int acc = 0;
        for (int b = 0; b < B_; b++) { int s = acc; acc += rl[b]; if (n < acc) { src = s; break; } }
    }

    for (int i = t; i < C_*C_; i += 256) {
        int o = i >> 6;
        float scale = bn[o] * rsqrtf(bn[3*C_ + o] + 1e-5f);
        Wl[i] = f2bf(w[i] * scale);
    }
    const float* xs = x + (size_t)src * C_ * HW_;
    for (int i = t; i < 128*C_; i += 256) {
        int c = i >> 7, pl = i & 127;
        Xl[pl*C_ + c] = f2bf(xs[(size_t)c*HW_ + p0 + pl]);
    }
    __syncthreads();

    int wave = t >> 5, lane = t & 31, half = lane >> 4, l15 = lane & 15, kb = half * 8;
    int pw = wave * 16;

    ABf bf0, bf1;
    load_a16(bf0, Xl + (pw + l15)*C_, 0  + kb);
    load_a16(bf1, Xl + (pw + l15)*C_, 32 + kb);

    #pragma unroll
    for (int mt = 0; mt < 4; mt++) {
        v8f acc = vzero();
        ABf af;
        load_a16(af, Wl + (mt*16 + l15)*C_, 0 + kb);
        acc = wmma_bf16(af, bf0, acc);
        load_a16(af, Wl + (mt*16 + l15)*C_, 32 + kb);
        acc = wmma_bf16(af, bf1, acc);

        int p = p0 + pw + l15;
        #pragma unroll
        for (int r = 0; r < 8; r++) {
            int o = mt*16 + r + 8*half;
            float scale = bn[o] * rsqrtf(bn[3*C_ + o] + 1e-5f);
            float beff  = (bias[o] - bn[2*C_ + o]) * scale + bn[C_ + o];
            float v = acc[r] + beff;
            out[(size_t)(n*C_ + o)*HW_ + p] = f2bf(v > 0.f ? v : 0.f);
        }
    }
}

// ---------------------------------------------------------------------------
// Kernel B/C: criss-cross attention partials along one axis.
// COL:  fixed w = blockIdx.x, rows/cols are h/g (diag masked)
// !COL: fixed h = blockIdx.x, rows/cols are w/v
// Emits per-row streaming-softmax partials m,s and U = exp(E-m) * V^T (bf16).
// ---------------------------------------------------------------------------
template<bool COL>
__global__ __launch_bounds__(256) void attn_kernel(const unsigned short* __restrict__ pq,
                                                   const unsigned short* __restrict__ pk,
                                                   const unsigned short* __restrict__ pv,
                                                   float* __restrict__ mOut,
                                                   float* __restrict__ sOut,
                                                   unsigned short* __restrict__ uOut) {
    __shared__ __align__(16) unsigned short Qt[128*C_];   // [s][c]
    __shared__ __align__(16) unsigned short Kt[128*C_];   // [s][c]
    __shared__ __align__(16) unsigned short Vtt[C_*128];  // [c][s]
    __shared__ __align__(16) unsigned short Pl[8*16*128]; // per-wave exp(E-m) tiles

    int n = blockIdx.y;
    int fix = blockIdx.x;
    int t = threadIdx.x;
    size_t basep = (size_t)n * C_ * HW_;

    for (int i = t; i < 128*C_; i += 256) {
        int c = i >> 7, s = i & 127;
        size_t g = basep + (size_t)c*HW_ + (COL ? (s*W_ + fix) : (fix*W_ + s));
        Qt[s*C_ + c]  = pq[g];
        Kt[s*C_ + c]  = pk[g];
        Vtt[c*128 + s] = pv[g];
    }
    __syncthreads();

    int wave = t >> 5, lane = t & 31, half = lane >> 4, l15 = lane & 15, kb = half * 8;
    int h0 = wave * 16;

    ABf af0, af1;
    load_a16(af0, Qt + (h0 + l15)*C_, 0  + kb);
    load_a16(af1, Qt + (h0 + l15)*C_, 32 + kb);

    v8f acc[8];
    #pragma unroll
    for (int gt = 0; gt < 8; gt++) {
        acc[gt] = vzero();
        ABf bfr;
        load_a16(bfr, Kt + (gt*16 + l15)*C_, 0 + kb);
        acc[gt] = wmma_bf16(af0, bfr, acc[gt]);
        load_a16(bfr, Kt + (gt*16 + l15)*C_, 32 + kb);
        acc[gt] = wmma_bf16(af1, bfr, acc[gt]);
    }

    float mrow[8], srow[8];
    #pragma unroll
    for (int r = 0; r < 8; r++) {
        int hg = h0 + r + 8*half;
        if (COL) {
            int gtd = hg >> 4;
            if ((hg & 15) == l15) acc[gtd][r] = NEG_;
        }
        float mx = NEG_;
        #pragma unroll
        for (int gt = 0; gt < 8; gt++) mx = fmaxf(mx, acc[gt][r]);
        for (int m = 1; m < 16; m <<= 1) mx = fmaxf(mx, __shfl_xor(mx, m, 32));
        float s = 0.f;
        #pragma unroll
        for (int gt = 0; gt < 8; gt++) { float e = __expf(acc[gt][r] - mx); acc[gt][r] = e; s += e; }
        for (int m = 1; m < 16; m <<= 1) s += __shfl_xor(s, m, 32);
        mrow[r] = mx; srow[r] = s;
    }

    if (l15 == 0) {
        #pragma unroll
        for (int r = 0; r < 8; r++) {
            int hg = h0 + r + 8*half;
            size_t oi = COL ? ((size_t)(n*H_ + hg)*W_ + fix)
                            : ((size_t)(n*H_ + fix)*W_ + hg);
            mOut[oi] = mrow[r];
            sOut[oi] = srow[r];
        }
    }

    unsigned short* Pw = Pl + wave * 16 * 128;
    #pragma unroll
    for (int gt = 0; gt < 8; gt++)
        #pragma unroll
        for (int r = 0; r < 8; r++)
            Pw[(r + 8*half)*128 + gt*16 + l15] = f2bf(acc[gt][r]);
    __syncthreads();

    // U[h][c] = sum_g P[h][g] * V[c][g]
    v8f acc2[4];
    #pragma unroll
    for (int ct = 0; ct < 4; ct++) acc2[ct] = vzero();
    #pragma unroll
    for (int ks = 0; ks < 4; ks++) {
        ABf a2;
        load_a16(a2, Pw + l15*128, ks*32 + kb);
        #pragma unroll
        for (int ct = 0; ct < 4; ct++) {
            ABf b2;
            load_a16(b2, Vtt + (ct*16 + l15)*128, ks*32 + kb);
            acc2[ct] = wmma_bf16(a2, b2, acc2[ct]);
        }
    }
    #pragma unroll
    for (int ct = 0; ct < 4; ct++)
        #pragma unroll
        for (int r = 0; r < 8; r++) {
            int hg = h0 + r + 8*half;
            int c  = ct*16 + l15;
            size_t gi = basep + (size_t)c*HW_ + (COL ? (hg*W_ + fix) : (fix*W_ + hg));
            uOut[gi] = f2bf(acc2[ct][r]);
        }
}

// ---------------------------------------------------------------------------
// Kernel D: merge the two streaming-softmax partials into scale factors.
// ---------------------------------------------------------------------------
__global__ void fac_kernel(const float* __restrict__ mH, const float* __restrict__ sH,
                           const float* __restrict__ mW, const float* __restrict__ sW,
                           const float* __restrict__ gamma,
                           float* __restrict__ facH, float* __restrict__ facW) {
    int i = blockIdx.x * 256 + threadIdx.x;
    if (i >= N_*HW_) return;
    float a = mH[i], b = mW[i];
    float M = fmaxf(a, b);
    float eh = __expf(a - M), ew = __expf(b - M);
    float Z = sH[i]*eh + sW[i]*ew;
    float g = gamma[0] / Z;
    facH[i] = g * eh;
    facW[i] = g * ew;
}

// ---------------------------------------------------------------------------
// Kernel E: cc = facH*uH + facW*uW + x;  per-batch prefix max+mean over n.
// ---------------------------------------------------------------------------
__global__ void fuse_kernel(const float* __restrict__ x, const int* __restrict__ rl,
                            const float* __restrict__ facH, const float* __restrict__ facW,
                            const unsigned short* __restrict__ uH,
                            const unsigned short* __restrict__ uW,
                            unsigned short* __restrict__ fused) {
    int i = blockIdx.x * 256 + threadIdx.x;    // c*HW + p
    if (i >= C_*HW_) return;
    int c = i / HW_, p = i - c*HW_;
    float mx = NEG_, sm = 0.f;
    int b = 0;
    int cumb = rl[0];
    for (int n = 0; n < N_; n++) {
        float fH = facH[n*HW_ + p], fW = facW[n*HW_ + p];
        size_t gi = (size_t)(n*C_ + c)*HW_ + p;
        float cc = fH * bf2f(uH[gi]) + fW * bf2f(uW[gi]) + x[gi];
        mx = fmaxf(mx, cc);
        sm += cc;
        if (n + 1 == cumb && b < B_) {
            fused[(size_t)(b*C_ + c)*HW_ + p] = f2bf(mx + sm / (float)(n + 1));
            b++;
            if (b < B_) cumb += rl[b];
        }
    }
}

// ---------------------------------------------------------------------------
// Kernel F: 3x3 SAME conv + bias + BN + ReLU via 9-tap implicit GEMM.
// grid = (H, B), block = 256; wave handles 16 output cols, all 64 out chans.
// ---------------------------------------------------------------------------
__global__ __launch_bounds__(256) void conv_kernel(const unsigned short* __restrict__ fused,
                                                   const float* __restrict__ cw,
                                                   const float* __restrict__ cb,
                                                   const float* __restrict__ bn,
                                                   float* __restrict__ out) {
    __shared__ __align__(16) unsigned short WT[9*C_*C_];   // [tap][o][c], BN-scaled
    __shared__ __align__(16) unsigned short IN[3*C_*132];  // [row][c][wl], wl = w+1

    int h = blockIdx.x, b = blockIdx.y;
    int t = threadIdx.x;

    for (int i = t; i < 9*C_*C_; i += 256) {
        int tap = i / (C_*C_);
        int rem = i - tap*C_*C_;
        int o = rem >> 6, c = rem & 63;
        float scale = bn[o] * rsqrtf(bn[3*C_ + o] + 1e-5f);
        WT[i] = f2bf(cw[(size_t)(o*C_ + c)*9 + tap] * scale);
    }
    for (int i = t; i < 3*C_*132; i += 256) {
        int rr = i / (C_*132);
        int rem = i - rr*C_*132;
        int c = rem / 132, wl = rem - c*132;
        int hh = h + rr - 1, ww = wl - 1;
        unsigned short v = 0;
        if (hh >= 0 && hh < H_ && ww >= 0 && ww < W_ && wl < 130)
            v = fused[(size_t)(b*C_ + c)*HW_ + hh*W_ + ww];
        IN[i] = v;
    }
    __syncthreads();

    int wave = t >> 5, lane = t & 31, half = lane >> 4, l15 = lane & 15, kb = half * 8;
    int w0 = wave * 16;

    v8f acc[4];
    #pragma unroll
    for (int mt = 0; mt < 4; mt++) acc[mt] = vzero();

    #pragma unroll
    for (int tap = 0; tap < 9; tap++) {
        int rr = tap / 3, dx = tap % 3;
        #pragma unroll
        for (int ks = 0; ks < 2; ks++) {
            ABf bfr;
            int wl = w0 + l15 + dx;
            #pragma unroll
            for (int j = 0; j < 8; j++) {
                bfr.u[j]     = IN[(rr*C_ + ks*32 + kb + j)*132 + wl];
                bfr.u[8 + j] = IN[(rr*C_ + ks*32 + kb + 16 + j)*132 + wl];
            }
            #pragma unroll
            for (int mt = 0; mt < 4; mt++) {
                ABf afr;
                load_a16(afr, WT + (tap*C_ + mt*16 + l15)*C_, ks*32 + kb);
                acc[mt] = wmma_bf16(afr, bfr, acc[mt]);
            }
        }
    }

    #pragma unroll
    for (int mt = 0; mt < 4; mt++)
        #pragma unroll
        for (int r = 0; r < 8; r++) {
            int o = mt*16 + r + 8*half;
            float scale = bn[o] * rsqrtf(bn[3*C_ + o] + 1e-5f);
            float beff  = (cb[o] - bn[2*C_ + o]) * scale + bn[C_ + o];
            float v = acc[mt][r] + beff;
            out[((size_t)(b*C_ + o)*H_ + h)*W_ + w0 + l15] = fmaxf(v, 0.f);
        }
}

// ---------------------------------------------------------------------------
extern "C" void kernel_launch(void* const* d_in, const int* in_sizes, int n_in,
                              void* d_out, int out_size, void* d_ws, size_t ws_size,
                              hipStream_t stream) {
    const float* x     = (const float*)d_in[0];
    const int*   rl    = (const int*)  d_in[1];
    const float* q_w   = (const float*)d_in[2];
    const float* q_b   = (const float*)d_in[3];
    const float* q_bn  = (const float*)d_in[4];
    const float* k_w   = (const float*)d_in[5];
    const float* k_b   = (const float*)d_in[6];
    const float* k_bn  = (const float*)d_in[7];
    const float* v_w   = (const float*)d_in[8];
    const float* v_b   = (const float*)d_in[9];
    const float* v_bn  = (const float*)d_in[10];
    const float* gamma = (const float*)d_in[11];
    const float* cov_w = (const float*)d_in[12];
    const float* cov_b = (const float*)d_in[13];
    const float* cov_bn= (const float*)d_in[14];
    float* out = (float*)d_out;

    char* p = (char*)d_ws;
    auto alloc = [&](size_t bytes) { char* r = p; p += (bytes + 255) & ~(size_t)255; return r; };

    const size_t E = (size_t)N_ * C_ * HW_;
    unsigned short* pq    = (unsigned short*)alloc(E * 2);
    unsigned short* pk    = (unsigned short*)alloc(E * 2);
    unsigned short* pv    = (unsigned short*)alloc(E * 2);
    unsigned short* uH    = (unsigned short*)alloc(E * 2);
    unsigned short* uW    = (unsigned short*)alloc(E * 2);
    float* mH   = (float*)alloc((size_t)N_*HW_*4);
    float* sH   = (float*)alloc((size_t)N_*HW_*4);
    float* mW   = (float*)alloc((size_t)N_*HW_*4);
    float* sW   = (float*)alloc((size_t)N_*HW_*4);
    float* facH = (float*)alloc((size_t)N_*HW_*4);
    float* facW = (float*)alloc((size_t)N_*HW_*4);
    unsigned short* fusedB = (unsigned short*)alloc((size_t)B_*C_*HW_*2);

    dim3 blk(256);
    cbr_kernel<<<dim3(HW_/128, N_), blk, 0, stream>>>(x, rl, q_w, q_b, q_bn, pq, 1);
    cbr_kernel<<<dim3(HW_/128, N_), blk, 0, stream>>>(x, rl, k_w, k_b, k_bn, pk, 0);
    cbr_kernel<<<dim3(HW_/128, N_), blk, 0, stream>>>(x, rl, v_w, v_b, v_bn, pv, 0);

    attn_kernel<true ><<<dim3(W_, N_), blk, 0, stream>>>(pq, pk, pv, mH, sH, uH);
    attn_kernel<false><<<dim3(H_, N_), blk, 0, stream>>>(pq, pk, pv, mW, sW, uW);

    fac_kernel<<<dim3((N_*HW_ + 255)/256), blk, 0, stream>>>(mH, sH, mW, sW, gamma, facH, facW);
    fuse_kernel<<<dim3((C_*HW_ + 255)/256), blk, 0, stream>>>(x, rl, facH, facW, uH, uW, fusedB);

    conv_kernel<<<dim3(H_, B_), blk, 0, stream>>>(fusedB, cov_w, cov_b, cov_bn, out);
}
